// CTSelfAttention_66614942761212
// MI455X (gfx1250) — compile-verified, matrix operand order checked by reading, not run on hardware
//
#include <hip/hip_runtime.h>
#include <hip/hip_bf16.h>

#define S_LEN 2048
#define D_MODEL 1024
#define N_HEADS 16
#define HEAD_DIM 64
#define MAX_FUT 128

typedef __attribute__((ext_vector_type(16))) __bf16   v16bf;
typedef __attribute__((ext_vector_type(8)))  float    v8f;
typedef __attribute__((ext_vector_type(4)))  unsigned v4u;
typedef __attribute__((ext_vector_type(8)))  unsigned v8u;
typedef __attribute__((ext_vector_type(4)))  unsigned u32x4;
typedef __attribute__((ext_vector_type(4)))  int      i32x4;
typedef __attribute__((ext_vector_type(8)))  int      i32x8;

union BFrag { v8u u; v16bf b; };

__device__ __forceinline__ unsigned short f2bf(float f) {
    union { float f; unsigned u; } v; v.f = f;
    unsigned r = v.u + 0x7FFFu + ((v.u >> 16) & 1u);
    return (unsigned short)(r >> 16);
}

// Load one 16x32 bf16 WMMA operand fragment for this lane from a row-major
// bf16 matrix (works for global or LDS pointers).
// Layout (CDNA5 ISA, 16-bit A-matrix 16x32): lane&15 = row M,
// khalf = lane>>4: elements 0..7 <- K = khalf*8+0..7, 8..15 <- khalf*8+16..23.
// Both runs are 8 contiguous bf16 (16B) -> two b128 loads.
__device__ __forceinline__ v16bf load_frag(const unsigned short* base,
                                           int row_stride, int row,
                                           int kbase, int khalf) {
    const unsigned short* p = base + (size_t)row * row_stride + kbase + khalf * 8;
    v4u lo = *reinterpret_cast<const v4u*>(p);
    v4u hi = *reinterpret_cast<const v4u*>(p + 16);
    BFrag f;
#pragma unroll
    for (int i = 0; i < 4; ++i) { f.u[i] = lo[i]; f.u[i + 4] = hi[i]; }
    return f.b;
}

__device__ __forceinline__ v8f wmma_bf16(v16bf a, v16bf b, v8f c) {
    return __builtin_amdgcn_wmma_f32_16x16x32_bf16(
        false, a, false, b, (short)0, c, false, false);
}

// Issue a TDM 1-D copy of one 32-key K chunk (32 tokens x 64 dh bf16 = 4KB,
// contiguous) from global memory into this wave's LDS staging buffer.
// D# per CDNA5 ISA; tracked by TENSORcnt.
__device__ __forceinline__ void tdm_load_k_chunk(const unsigned short* gsrc,
                                                 unsigned short* ldst) {
    unsigned long long ga = (unsigned long long)(uintptr_t)gsrc;
    unsigned galo = __builtin_amdgcn_readfirstlane((unsigned)ga);
    unsigned gahi = __builtin_amdgcn_readfirstlane((unsigned)(ga >> 32));
    unsigned loff = __builtin_amdgcn_readfirstlane((unsigned)(uintptr_t)ldst);

    u32x4 g0 = { 1u,                                    // count=1, user mode
                 loff,                                  // lds_addr (bytes)
                 galo,                                  // global_addr[31:0]
                 (gahi & 0x01FFFFFFu) | 0x80000000u };  // addr[56:32] | type=2
    i32x8 g1 = { (int)(1u << 16),   // data_size=1 -> 2-byte elements
                 (int)0x08000000u,  // tensor_dim0[15:0] = 2048 @ bits 63:48
                 (int)0x00010000u,  // tensor_dim0 hi = 0, tensor_dim1 = 1
                 (int)0x08000000u,  // tile_dim0 = 2048 @ bits 127:112
                 1,                 // tile_dim1 = 1
                 2048,              // tensor_dim0_stride
                 0, 0 };
    i32x4 z4 = {0, 0, 0, 0};
#if __clang_major__ >= 23
    i32x8 z8 = {0, 0, 0, 0, 0, 0, 0, 0};
    __builtin_amdgcn_tensor_load_to_lds(g0, g1, z4, z4, z8, 0);
#else
    __builtin_amdgcn_tensor_load_to_lds(g0, g1, z4, z4, 0);
#endif
}

// ---------------------------------------------------------------- convert ---
__global__ __launch_bounds__(256) void cvt_bf16(const float* __restrict__ in,
                                                unsigned short* __restrict__ out,
                                                int n) {
    int i = blockIdx.x * 256 + threadIdx.x;
    if (i < n) out[i] = f2bf(in[i]);
}

// -------------------------------------------------------------- QKV GEMM ----
// out[m,n] = sum_k X[m,k] * W[n,k] + bias[n]   (hidden @ W.T + b)
// One wave per 16x16 tile. mat 0->Q, 1->K ([B][H][S][Dh]), 2->V transposed
// [B][H][Dh][S] so the attention P@V B-fragments are contiguous.
__global__ __launch_bounds__(256) void qkv_gemm(
    const unsigned short* __restrict__ Xb,
    const unsigned short* __restrict__ Wb,
    const float* __restrict__ bq, const float* __restrict__ bk,
    const float* __restrict__ bv,
    unsigned short* __restrict__ Qb, unsigned short* __restrict__ Kb,
    unsigned short* __restrict__ Vtb) {
    int lane = threadIdx.x & 31;
    int wave = threadIdx.x >> 5;
    int t = blockIdx.x * 8 + wave;          // 0..49151
    int mat = t / 16384;
    int r = t - mat * 16384;
    int m0 = (r >> 6) * 16;
    int n0 = (r & 63) * 16;
    const unsigned short* W = Wb + (size_t)mat * D_MODEL * D_MODEL;

    int ln = lane & 15, khalf = lane >> 4;
    int arow = m0 + ln;
    int brow = n0 + ln;

    v8f acc = {0.f, 0.f, 0.f, 0.f, 0.f, 0.f, 0.f, 0.f};
#pragma unroll 4
    for (int k0 = 0; k0 < D_MODEL; k0 += 32) {
        __builtin_prefetch(Xb + (size_t)arow * D_MODEL + k0 + 64, 0, 1);
        v16bf a = load_frag(Xb, D_MODEL, arow, k0, khalf);
        v16bf b = load_frag(W,  D_MODEL, brow, k0, khalf);
        acc = wmma_bf16(a, b, acc);
    }

    const float* bias = (mat == 0) ? bq : (mat == 1) ? bk : bv;
    float bn = bias[n0 + ln];
    int nglob = n0 + ln;
    int h = nglob >> 6, dh = nglob & 63;

#pragma unroll
    for (int rr = 0; rr < 8; ++rr) {
        int m = m0 + rr + 8 * khalf;        // C layout: M = vgpr + 8*(lane>=16)
        int bb = m >> 11, s = m & (S_LEN - 1);
        unsigned short bf = f2bf(acc[rr] + bn);
        size_t hd = (size_t)(bb * N_HEADS + h);
        if (mat == 0)      Qb[(hd * S_LEN + s) * HEAD_DIM + dh] = bf;
        else if (mat == 1) Kb[(hd * S_LEN + s) * HEAD_DIM + dh] = bf;
        else               Vtb[(hd * HEAD_DIM + dh) * S_LEN + s] = bf;
    }
}

// ------------------------------------------------------------- attention ----
// One wave per 16-row query tile. K band is staged into LDS with TDM
// (double-buffered 32-key chunks, gated by s_wait_tensorcnt) so the DMA
// overlaps the previous chunk's WMMAs + softmax VALU work.
//  pass1: score WMMAs from staged K, online row max/sum (shfl_xor halves)
//  pass2: recompute, write normalized probs (band + zero fill), relayout P
//         through per-wave LDS tile, P@V via WMMA against transposed V.
__global__ __launch_bounds__(128) void attn_banded(
    const unsigned short* __restrict__ Qb,
    const unsigned short* __restrict__ Kb,
    const unsigned short* __restrict__ Vtb,
    const int* __restrict__ amask,
    float* __restrict__ ctx, float* __restrict__ probs) {
    // per wave: 2 x 2048 ushort K staging buffers (4KB each) + 512 ushort P tile
    __shared__ unsigned short smem[4 * 4608];

    int lane = threadIdx.x & 31;
    int wave = threadIdx.x >> 5;
    int ln = lane & 15, khalf = lane >> 4;
    int qt = blockIdx.x * 4 + wave;               // query tile 0..127
    int h = blockIdx.y, b = blockIdx.z;
    int q0 = qt * 16;

    unsigned short* stage = smem + wave * 4608;         // two 2048-ushort buffers
    unsigned short* ptile = smem + wave * 4608 + 4096;  // 16x32 P relayout tile

    const unsigned short* Qh = Qb  + ((size_t)(b * N_HEADS + h) * S_LEN) * HEAD_DIM;
    const unsigned short* Kh = Kb  + ((size_t)(b * N_HEADS + h) * S_LEN) * HEAD_DIM;
    const unsigned short* Vh = Vtb + ((size_t)(b * N_HEADS + h) * HEAD_DIM) * S_LEN;
    float* probsRow = probs + (size_t)(b * N_HEADS + h) * S_LEN * S_LEN;

    v16bf qa0 = load_frag(Qh, HEAD_DIM, q0 + ln, 0, khalf);
    v16bf qa1 = load_frag(Qh, HEAD_DIM, q0 + ln, 32, khalf);

    int jmax = q0 + 15 + MAX_FUT;
    if (jmax > S_LEN - 1) jmax = S_LEN - 1;
    int lastTile = jmax >> 4;                     // inclusive
    int lastChunk = lastTile >> 1;                // 32-key chunks, inclusive

    float m[8], l[8];
#pragma unroll
    for (int rr = 0; rr < 8; ++rr) { m[rr] = -3.0e38f; l[rr] = 0.f; }

    // ---- pass 1: softmax statistics over the band (TDM-staged K)
    tdm_load_k_chunk(Kh, stage);                  // chunk 0 -> buffer 0
    for (int ch = 0; ch <= lastChunk; ++ch) {
        if (ch < lastChunk) {
            tdm_load_k_chunk(Kh + (size_t)(ch + 1) * 32 * HEAD_DIM,
                             stage + ((ch + 1) & 1) * 2048);
            __builtin_amdgcn_s_wait_tensorcnt(1); // oldest chunk landed
        } else {
            __builtin_amdgcn_s_wait_tensorcnt(0); // drain
        }
        const unsigned short* kbuf = stage + (ch & 1) * 2048;
#pragma unroll
        for (int sub = 0; sub < 2; ++sub) {
            int kt = ch * 2 + sub;
            v16bf kb0 = load_frag(kbuf, HEAD_DIM, sub * 16 + ln, 0, khalf);
            v16bf kb1 = load_frag(kbuf, HEAD_DIM, sub * 16 + ln, 32, khalf);
            v8f c = {0.f, 0.f, 0.f, 0.f, 0.f, 0.f, 0.f, 0.f};
            c = wmma_bf16(qa0, kb0, c);
            c = wmma_bf16(qa1, kb1, c);
            int j = kt * 16 + ln;
            bool jok = (kt <= lastTile) && (amask[b * S_LEN + j] != 0);
#pragma unroll
            for (int rr = 0; rr < 8; ++rr) {
                int i = q0 + rr + 8 * khalf;
                float s = (jok && j <= i + MAX_FUT) ? c[rr] : -3.0e38f;
                float tmax = s;
                for (int off = 1; off < 16; off <<= 1)
                    tmax = fmaxf(tmax, __shfl_xor(tmax, off, 32));
                float nm = fmaxf(m[rr], tmax);
                float e = (s > -1.0e37f) ? __expf((s - nm) * 0.125f) : 0.f;
                float tsum = e;
                for (int off = 1; off < 16; off <<= 1)
                    tsum += __shfl_xor(tsum, off, 32);
                float scale = (m[rr] > -1.0e37f) ? __expf((m[rr] - nm) * 0.125f) : 0.f;
                l[rr] = l[rr] * scale + tsum;
                m[rr] = nm;
            }
        }
    }
    float rl[8];
#pragma unroll
    for (int rr = 0; rr < 8; ++rr) rl[rr] = (l[rr] > 0.f) ? 1.0f / l[rr] : 0.f;

    // ---- pass 2: probs + ctx (TDM-staged K again; V direct from L2)
    v8f cc[4];
#pragma unroll
    for (int nd = 0; nd < 4; ++nd)
        cc[nd] = (v8f){0.f, 0.f, 0.f, 0.f, 0.f, 0.f, 0.f, 0.f};

    tdm_load_k_chunk(Kh, stage);
    for (int ch = 0; ch <= lastChunk; ++ch) {
        if (ch < lastChunk) {
            tdm_load_k_chunk(Kh + (size_t)(ch + 1) * 32 * HEAD_DIM,
                             stage + ((ch + 1) & 1) * 2048);
            __builtin_amdgcn_s_wait_tensorcnt(1);
        } else {
            __builtin_amdgcn_s_wait_tensorcnt(0);
        }
        const unsigned short* kbuf = stage + (ch & 1) * 2048;
#pragma unroll
        for (int sub = 0; sub < 2; ++sub) {
            int kt = ch * 2 + sub;
            v16bf kb0 = load_frag(kbuf, HEAD_DIM, sub * 16 + ln, 0, khalf);
            v16bf kb1 = load_frag(kbuf, HEAD_DIM, sub * 16 + ln, 32, khalf);
            v8f c = {0.f, 0.f, 0.f, 0.f, 0.f, 0.f, 0.f, 0.f};
            c = wmma_bf16(qa0, kb0, c);
            c = wmma_bf16(qa1, kb1, c);
            int j = kt * 16 + ln;
            bool jok = (kt <= lastTile) && (amask[b * S_LEN + j] != 0);
#pragma unroll
            for (int rr = 0; rr < 8; ++rr) {
                int i = q0 + rr + 8 * khalf;
                float p = (jok && j <= i + MAX_FUT)
                              ? __expf((c[rr] - m[rr]) * 0.125f) * rl[rr]
                              : 0.f;
                probsRow[(size_t)i * S_LEN + j] = p;
                ptile[(rr + 8 * khalf) * 32 + sub * 16 + ln] = f2bf(p);
            }
        }
        // LDS relayout stores (cross-lane, same wave) must land before reads
        asm volatile("s_wait_dscnt 0" ::: "memory");
        v16bf pa = load_frag(ptile, 32, ln, 0, khalf);   // 16x32 P fragment
#pragma unroll
        for (int nd = 0; nd < 4; ++nd) {
            v16bf vb = load_frag(Vh, S_LEN, nd * 16 + ln, ch * 32, khalf);
            cc[nd] = wmma_bf16(pa, vb, cc[nd]);
        }
        asm volatile("s_wait_dscnt 0" ::: "memory");     // reads done before overwrite
    }

    // zero-fill probs beyond the band (every element written exactly once)
    for (int kt = (lastChunk + 1) * 2; kt < S_LEN / 16; ++kt) {
        int j = kt * 16 + ln;
#pragma unroll
        for (int rr = 0; rr < 8; ++rr)
            probsRow[(size_t)(q0 + rr + 8 * khalf) * S_LEN + j] = 0.f;
    }

    // ctx out: [B][S][D] f32
#pragma unroll
    for (int nd = 0; nd < 4; ++nd) {
#pragma unroll
        for (int rr = 0; rr < 8; ++rr) {
            int i = q0 + rr + 8 * khalf;
            ctx[(size_t)(b * S_LEN + i) * D_MODEL + h * HEAD_DIM + nd * 16 + ln] =
                cc[nd][rr];
        }
    }
}

// ---------------------------------------------------------------------------
extern "C" void kernel_launch(void* const* d_in, const int* in_sizes, int n_in,
                              void* d_out, int out_size, void* d_ws, size_t ws_size,
                              hipStream_t stream) {
    (void)in_sizes; (void)n_in; (void)out_size; (void)ws_size;
    const float* hidden = (const float*)d_in[0];
    const int*   amask  = (const int*)d_in[1];
    const float* Wq     = (const float*)d_in[2];
    const float* bq     = (const float*)d_in[3];
    const float* Wk     = (const float*)d_in[4];
    const float* bk     = (const float*)d_in[5];
    const float* Wv     = (const float*)d_in[6];
    const float* bv     = (const float*)d_in[7];

    float* out   = (float*)d_out;
    float* ctx   = out;                                   // [2][2048][1024]
    float* probs = out + (size_t)2 * S_LEN * D_MODEL;     // [2][16][2048][2048]

    const size_t NX = (size_t)2 * S_LEN * D_MODEL;        // 4,194,304
    const size_t NW = (size_t)D_MODEL * D_MODEL;          // 1,048,576
    unsigned short* Xb  = (unsigned short*)d_ws;
    unsigned short* Wb  = Xb + NX;                        // 3 mats
    unsigned short* Qb  = Wb + 3 * NW;
    unsigned short* Kb  = Qb + NX;
    unsigned short* Vtb = Kb + NX;

    cvt_bf16<<<(int)(NX / 256), 256, 0, stream>>>(hidden, Xb, (int)NX);
    cvt_bf16<<<(int)(NW / 256), 256, 0, stream>>>(Wq, Wb,          (int)NW);
    cvt_bf16<<<(int)(NW / 256), 256, 0, stream>>>(Wk, Wb + NW,     (int)NW);
    cvt_bf16<<<(int)(NW / 256), 256, 0, stream>>>(Wv, Wb + 2 * NW, (int)NW);

    qkv_gemm<<<6144, 256, 0, stream>>>(Xb, Wb, bq, bk, bv, Qb, Kb, Vtb);

    attn_banded<<<dim3(32, N_HEADS, 2), 128, 0, stream>>>(Qb, Kb, Vtb, amask,
                                                          ctx, probs);
}
